// GConv_37417755083211
// MI455X (gfx1250) — compile-verified, matrix operand order checked by reading, not run on hardware
//
#include <hip/hip_runtime.h>

#define D 128

typedef __attribute__((ext_vector_type(16))) __bf16 v16bf;
typedef __attribute__((ext_vector_type(8)))  float  v8f;

// ---------- bf16 helpers: native hardware converts on gfx1250 ----------
static __device__ __forceinline__ __bf16 f2bf(float f) { return (__bf16)f; }
static __device__ __forceinline__ float  bf2f(__bf16 b) { return (float)b; }

// ---------- memset ----------
__global__ void zero_kernel(float* __restrict__ p, long long n) {
  long long i = (long long)blockIdx.x * blockDim.x + threadIdx.x;
  long long stride = (long long)gridDim.x * blockDim.x;
  for (; i < n; i += stride) p[i] = 0.0f;
}

// ---------- out-degree ----------
__global__ void degree_kernel(const int* __restrict__ src, float* __restrict__ deg, int E) {
  int e = blockIdx.x * blockDim.x + threadIdx.x;
  if (e < E) unsafeAtomicAdd(&deg[src[e]], 1.0f);
}

__global__ void invdeg_kernel(float* __restrict__ deg, int N) {
  int i = blockIdx.x * blockDim.x + threadIdx.x;
  if (i < N) deg[i] = 1.0f / fmaxf(deg[i], 1.0f);
}

// ---------- SpMM scatter-add: agg[dst] += h[src] * inv[src] ----------
// One thread handles one (edge, 4-feature chunk): wave32 covers a full 128-f row.
__global__ void scatter_kernel(const float* __restrict__ h, const float* __restrict__ inv,
                               const int* __restrict__ src, const int* __restrict__ dst,
                               float* __restrict__ agg, int E) {
  long long t = (long long)blockIdx.x * blockDim.x + threadIdx.x;
  int e = (int)(t >> 5);
  int c = (int)(t & 31);
  if (e >= E) return;
  int s = src[e];
  int d = dst[e];
  float sc = inv[s];
  float4 v = ((const float4*)(h + (size_t)s * D))[c];
  float* o = agg + (size_t)d * D + c * 4;
  unsafeAtomicAdd(o + 0, v.x * sc);
  unsafeAtomicAdd(o + 1, v.y * sc);
  unsafeAtomicAdd(o + 2, v.z * sc);
  unsafeAtomicAdd(o + 3, v.w * sc);
}

// ---------- GEMM + bias + PReLU via WMMA bf16x3 split precision ----------
// out[N,128] = prelu(agg[N,128] @ W[128,128] + b)
// Block = 256 threads = 8 waves. Each wave owns a 16-row strip, all 8 column tiles.
// W is staged in LDS pre-swizzled into exact 16x16x32 bf16 B-fragment layout:
//   fragment id f = kt*8+nt (kt in 0..3, nt in 0..7); lane holds 16 contiguous bf16.
// B element mapping per lane: n = nt*16 + (lane&15), half = lane>>4,
//   e<8  -> k = kt*32 + half*8 + e
//   e>=8 -> k = kt*32 + 16 + half*8 + (e-8)
__global__ __launch_bounds__(256) void gemm_prelu_kernel(
    const float* __restrict__ agg, const float* __restrict__ W,
    const float* __restrict__ bias, const float* __restrict__ prelu_a,
    float* __restrict__ out, int N) {
  __shared__ v16bf sWv[2048];              // 1024 hi fragments' lanes + 1024 lo (64 KB)
  __bf16* sW = (__bf16*)sWv;

  const int tid = threadIdx.x;
  // ---- stage W -> LDS as hi/lo bf16 fragments ----
  for (int idx = tid; idx < 16384; idx += 256) {
    int e    = idx & 15;
    int lane = (idx >> 4) & 31;
    int nt   = (idx >> 9) & 7;
    int kt   = idx >> 12;
    int n    = nt * 16 + (lane & 15);
    int half = lane >> 4;
    int k    = kt * 32 + ((e < 8) ? (half * 8 + e) : (16 + half * 8 + (e - 8)));
    float w  = W[k * D + n];
    __bf16 hi = f2bf(w);
    sW[idx]         = hi;
    sW[16384 + idx] = f2bf(w - bf2f(hi));
  }
  __syncthreads();

  const int wave = tid >> 5;
  const int lane = tid & 31;
  const int half = lane >> 4;
  const int m    = lane & 15;
  const int rowBase = (blockIdx.x * 8 + wave) * 16;
  if (rowBase >= N) return;                // wave-uniform exit; EXEC stays all-1s below

  const float a = prelu_a[0];

  v8f acc[8];
#pragma unroll
  for (int nt = 0; nt < 8; ++nt) {
    float bv = bias[nt * 16 + m];          // bias depends on column only
#pragma unroll
    for (int v = 0; v < 8; ++v) acc[nt][v] = bv;
  }

  const float* aRow = agg + (size_t)(rowBase + m) * D;
#pragma unroll
  for (int kt = 0; kt < 4; ++kt) {
    // A-fragment: two contiguous 8-float runs per lane
    int k0 = kt * 32 + half * 8;
    float4 r0 = *(const float4*)(aRow + k0);
    float4 r1 = *(const float4*)(aRow + k0 + 4);
    float4 r2 = *(const float4*)(aRow + k0 + 16);
    float4 r3 = *(const float4*)(aRow + k0 + 20);
    float av[16] = {r0.x, r0.y, r0.z, r0.w, r1.x, r1.y, r1.z, r1.w,
                    r2.x, r2.y, r2.z, r2.w, r3.x, r3.y, r3.z, r3.w};
    v16bf a_hi, a_lo;
#pragma unroll
    for (int e = 0; e < 16; ++e) {
      __bf16 h = f2bf(av[e]);
      a_hi[e] = h;
      a_lo[e] = f2bf(av[e] - bf2f(h));
    }
#pragma unroll
    for (int nt = 0; nt < 8; ++nt) {
      int f = (kt * 8 + nt) * 32 + lane;
      v16bf b_hi = sWv[f];
      v16bf b_lo = sWv[1024 + f];
      acc[nt] = __builtin_amdgcn_wmma_f32_16x16x32_bf16(false, a_hi, false, b_hi,
                                                        (short)0, acc[nt], false, false);
      acc[nt] = __builtin_amdgcn_wmma_f32_16x16x32_bf16(false, a_lo, false, b_hi,
                                                        (short)0, acc[nt], false, false);
      acc[nt] = __builtin_amdgcn_wmma_f32_16x16x32_bf16(false, a_hi, false, b_lo,
                                                        (short)0, acc[nt], false, false);
    }
  }

  // ---- PReLU + store (C/D layout: VGPR v holds row m=v+8*half, col = nt*16 + lane%16)
#pragma unroll
  for (int nt = 0; nt < 8; ++nt) {
    int col = nt * 16 + m;
#pragma unroll
    for (int v = 0; v < 8; ++v) {
      float z = acc[nt][v];
      z = (z >= 0.0f) ? z : a * z;
      out[(size_t)(rowBase + v + 8 * half) * D + col] = z;
    }
  }
}

extern "C" void kernel_launch(void* const* d_in, const int* in_sizes, int n_in,
                              void* d_out, int out_size, void* d_ws, size_t ws_size,
                              hipStream_t stream) {
  const float* x   = (const float*)d_in[0];
  const int*   src = (const int*)d_in[1];
  const int*   dst = (const int*)d_in[2];
  const float* W1  = (const float*)d_in[3];
  const float* b1  = (const float*)d_in[4];
  const float* W2  = (const float*)d_in[5];
  const float* b2  = (const float*)d_in[6];
  const float* pa  = (const float*)d_in[7];
  float* out = (float*)d_out;

  const int N = in_sizes[0] / D;
  const int E = in_sizes[1];

  char* ws = (char*)d_ws;
  float* deg = (float*)ws;                                   // N floats -> becomes inv-degree
  size_t degBytes = (((size_t)N * sizeof(float)) + 255) & ~(size_t)255;
  float* agg = (float*)(ws + degBytes);                      // N*D floats
  const long long aggN = (long long)N * D;

  // degree + normalization factor
  zero_kernel<<<1024, 256, 0, stream>>>(deg, N);
  zero_kernel<<<4096, 256, 0, stream>>>(agg, aggN);
  degree_kernel<<<(E + 255) / 256, 256, 0, stream>>>(src, deg, E);
  invdeg_kernel<<<(N + 255) / 256, 256, 0, stream>>>(deg, N);

  const long long scThreads = (long long)E * 32;
  const int scBlocks = (int)((scThreads + 255) / 256);
  const int strips = (N + 15) / 16;
  const int gemmBlocks = (strips + 7) / 8;

  // layer 1: h1 (stored in d_out) = prelu(SpMM(x*inv) @ W1 + b1)
  scatter_kernel<<<scBlocks, 256, 0, stream>>>(x, deg, src, dst, agg, E);
  gemm_prelu_kernel<<<gemmBlocks, 256, 0, stream>>>(agg, W1, b1, pa, out, N);

  // layer 2: out = prelu(SpMM(h1*inv) @ W2 + b2)
  zero_kernel<<<4096, 256, 0, stream>>>(agg, aggN);
  scatter_kernel<<<scBlocks, 256, 0, stream>>>(out, deg, src, dst, agg, E);
  gemm_prelu_kernel<<<gemmBlocks, 256, 0, stream>>>(agg, W2, b2, pa, out, N);
}
